// BoltzformerDecoder_35270271435490
// MI455X (gfx1250) — matrix operand order repaired and checked by reading, not compiled, not run on hardware
//
#include <hip/hip_runtime.h>
#include <hip/hip_bf16.h>

typedef _Float16 f16;
typedef _Float16 v8h  __attribute__((ext_vector_type(8)));
typedef _Float16 v16h __attribute__((ext_vector_type(16)));
typedef float    v8f  __attribute__((ext_vector_type(8)));

#define DIM 256
#define QLEN 1024
#define BATCH 2
#define NHEADS 8
#define NSAMP 102   // int(1024 * 0.1)

union V16H { v16h v; v8h h[2]; };

__device__ __forceinline__ float sigmoidf_(float x) {
    return 1.0f / (1.0f + __expf(-x));
}

// ---------------------------------------------------------------------------
// Weight transpose + f32 -> f16 :  wt[n*256 + k] = (f16) w[k*256 + n]
// So wt row n holds B-matrix column n with contiguous K -> perfect B loads.
// ---------------------------------------------------------------------------
__global__ void k_wt(const float* __restrict__ w, f16* __restrict__ wt) {
    int idx = blockIdx.x * blockDim.x + threadIdx.x;   // 0 .. 65535
    int n = idx >> 8;
    int k = idx & 255;
    wt[n * DIM + k] = (f16)w[k * DIM + n];
}

// ---------------------------------------------------------------------------
// LayerNorm over D=256, one block per row, output f16
// ---------------------------------------------------------------------------
__global__ void k_ln(const float* __restrict__ x, const float* __restrict__ w,
                     const float* __restrict__ b, f16* __restrict__ out) {
    __shared__ float red[256];
    int row = blockIdx.x;          // 0 .. 2047
    int t   = threadIdx.x;         // 0 .. 255
    float v = x[(size_t)row * DIM + t];

    red[t] = v;
    __syncthreads();
    for (int s = 128; s > 0; s >>= 1) { if (t < s) red[t] += red[t + s]; __syncthreads(); }
    float mu = red[0] * (1.0f / DIM);
    __syncthreads();

    float d = v - mu;
    red[t] = d * d;
    __syncthreads();
    for (int s = 128; s > 0; s >>= 1) { if (t < s) red[t] += red[t + s]; __syncthreads(); }
    float var = red[0] * (1.0f / DIM);

    float y = d * rsqrtf(var + 1e-4f) * w[t] + b[t];
    out[(size_t)row * DIM + t] = (f16)y;
}

// ---------------------------------------------------------------------------
// One wave computes a 16x16 f32 tile = A(16xK) x B(Kx16), K = 256, f16 WMMA.
//   A: row-major f16, lane (L&15) = row m, halves per ISA A-layout
//   B: "column rows" row-major f16 (Brows + n*256 + k), lane (L&15) = col n
// ---------------------------------------------------------------------------
__device__ __forceinline__ v8f wmma_tile16x16(const f16* __restrict__ Arows,
                                              const f16* __restrict__ Brows,
                                              int lane) {
    const int half = lane >> 4;
    const int idx  = lane & 15;
    const f16* arow = Arows + idx * DIM;   // A row m = idx
    const f16* brow = Brows + idx * DIM;   // B col n = idx
    v8f c = {};
#pragma unroll
    for (int kk = 0; kk < DIM; kk += 32) {
        V16H a, b;
        a.h[0] = *(const v8h*)(arow + kk + 8 * half);        // K = kk + 8h .. +7
        a.h[1] = *(const v8h*)(arow + kk + 16 + 8 * half);   // K = kk+16+8h .. +7
        b.h[0] = *(const v8h*)(brow + kk + 8 * half);        // K = kk + 8h .. +7
        b.h[1] = *(const v8h*)(brow + kk + 16 + 8 * half);   // K = kk+16+8h .. +7
        c = __builtin_amdgcn_wmma_f32_16x16x32_f16(
                /*neg_a=*/false, a.v, /*neg_b=*/false, b.v,
                /*c_mod=*/(short)0, c, /*reuse_a=*/false, /*reuse_b=*/false);
    }
    return c;
}

// ---------------------------------------------------------------------------
// MLP GEMM: out[2048,256] = act(A[2048,256] @ W[256,256] + bias)
// One wave per 16x16 tile; 2048 tiles; 128-thread blocks (4 waves).
// ---------------------------------------------------------------------------
__global__ void k_gemm_mlp(const f16* __restrict__ A, const f16* __restrict__ WT,
                           const float* __restrict__ bias, f16* __restrict__ out,
                           int relu) {
    int wid  = (blockIdx.x * blockDim.x + threadIdx.x) >> 5;  // 0 .. 2047
    int lane = threadIdx.x & 31;
    int rt = wid >> 4;                 // 128 row tiles
    int ct = wid & 15;                 // 16 col tiles
    int rbase = rt * 16, cbase = ct * 16;

    v8f c = wmma_tile16x16(A + (size_t)rbase * DIM, WT + (size_t)cbase * DIM, lane);

    int half = lane >> 4, n = lane & 15;
    float bn = bias[cbase + n];
#pragma unroll
    for (int i = 0; i < 8; i++) {
        int m = i + 8 * half;                       // C layout: VGPR i, lane half
        float v = c[i] + bn;
        if (relu) v = fmaxf(v, 0.0f);
        out[(size_t)(rbase + m) * DIM + cbase + n] = (f16)v;
    }
}

// ---------------------------------------------------------------------------
// Scores: attn[b,q,k] = sigmoid( (me[b] @ me[b]^T)[q,k] / 16 )
// B operand of "me @ me^T" is just rows of me -> same loads as A.
// 2 batches * 64*64 tiles = 8192 waves.
// ---------------------------------------------------------------------------
__global__ void k_scores(const f16* __restrict__ me, float* __restrict__ attn) {
    int wid  = (blockIdx.x * blockDim.x + threadIdx.x) >> 5;  // 0 .. 8191
    int lane = threadIdx.x & 31;
    int b  = wid >> 12;
    int r  = wid & 4095;
    int qt = r >> 6, kt = r & 63;

    const f16* meb = me + (size_t)b * QLEN * DIM;
    v8f c = wmma_tile16x16(meb + (size_t)qt * 16 * DIM,
                           meb + (size_t)kt * 16 * DIM, lane);

    int half = lane >> 4, n = lane & 15;
    float* ab = attn + (size_t)b * QLEN * QLEN;
#pragma unroll
    for (int i = 0; i < 8; i++) {
        int m = i + 8 * half;
        float s = c[i] * (1.0f / 16.0f);            // 1/sqrt(256)
        ab[(size_t)(qt * 16 + m) * QLEN + kt * 16 + n] = sigmoidf_(s);
    }
}

// ---------------------------------------------------------------------------
// Row sums of Boltzmann numerator: rowsum[b,q] = sum_k (attn<0.5)*exp(attn*(2+lid))
// One block per (b,q) row; one float4 (b128) load per thread covers the row.
// ---------------------------------------------------------------------------
__global__ void k_rowsum(const float4* __restrict__ attn4, const int* __restrict__ lid,
                         float* __restrict__ rowsum) {
    __shared__ float red[256];
    int row = blockIdx.x;              // 0 .. 2047
    int t   = threadIdx.x;             // 0 .. 255, each owns 4 consecutive k
    float invtemp = 2.0f + (float)(*lid);   // temp = 1/(2+lid) => attn/temp = attn*(2+lid)

    float4 a4 = attn4[(size_t)row * (QLEN / 4) + t];
    float av[4] = {a4.x, a4.y, a4.z, a4.w};
    float acc = 0.0f;
#pragma unroll
    for (int j = 0; j < 4; j++)
        if (av[j] < 0.5f) acc += __expf(av[j] * invtemp);

    red[t] = acc;
    __syncthreads();
    for (int s = 128; s > 0; s >>= 1) { if (t < s) red[t] += red[t + s]; __syncthreads(); }
    if (t == 0) rowsum[row] = red[0];
}

// ---------------------------------------------------------------------------
// Final elementwise tail (dominant HBM kernel): per-(b,q,k) quantities computed
// once per 4-wide column group, applied to all 8 heads. All traffic is b128.
// 512K threads over [2, 1024, 256 x float4].
// ---------------------------------------------------------------------------
__global__ void k_final(const float4* __restrict__ attn4, const float* __restrict__ rowsum,
                        const float4* __restrict__ rnd4, const int* __restrict__ lid,
                        float4* __restrict__ out4) {
    int tid = blockIdx.x * blockDim.x + threadIdx.x;   // 0 .. 524287
    int b    = tid >> 18;                               // 256K float4 groups / batch
    int qk4  = tid & ((1 << 18) - 1);
    int q    = qk4 >> 8;                                // 256 groups per row

    float invtemp = 2.0f + (float)(*lid);
    float4 a4 = attn4[(size_t)b * (QLEN * QLEN / 4) + qk4];
    float rs  = rowsum[b * QLEN + q] + 1e-6f;

    float av[4] = {a4.x, a4.y, a4.z, a4.w};
    float mp[4], sm[4];
#pragma unroll
    for (int j = 0; j < 4; j++) {
        float a  = av[j];
        float e  = (a < 0.5f) ? __expf(a * invtemp) : 0.0f;
        float bp = e / rs;
        float omb = fmaxf(1.0f - bp, 0.0f);
        mp[j] = __powf(omb, (float)NSAMP);              // (1-bp)^102
        sm[j] = sigmoidf_((0.5f - a) * 100.0f);         // 1 - sigmoid((a-0.5)*100)
    }

    size_t base = (size_t)b * NHEADS * (QLEN * QLEN / 4) + qk4;
#pragma unroll
    for (int h = 0; h < NHEADS; h++) {
        size_t o = base + (size_t)h * (QLEN * QLEN / 4);
        float4 r = rnd4[o];
        float4 ov;
        ov.x = sm[0] * sigmoidf_((mp[0] - r.x) * 100.0f);  // 1 - sigmoid((rand-mp)*100)
        ov.y = sm[1] * sigmoidf_((mp[1] - r.y) * 100.0f);
        ov.z = sm[2] * sigmoidf_((mp[2] - r.z) * 100.0f);
        ov.w = sm[3] * sigmoidf_((mp[3] - r.w) * 100.0f);
        out4[o] = ov;
    }
}

// ---------------------------------------------------------------------------
extern "C" void kernel_launch(void* const* d_in, const int* in_sizes, int n_in,
                              void* d_out, int out_size, void* d_ws, size_t ws_size,
                              hipStream_t stream) {
    (void)in_sizes; (void)n_in; (void)out_size; (void)ws_size;

    const float* tgt = (const float*)d_in[0];
    const float* lnw = (const float*)d_in[1];
    const float* lnb = (const float*)d_in[2];
    const float* w1  = (const float*)d_in[3];
    const float* b1  = (const float*)d_in[4];
    const float* w2  = (const float*)d_in[5];
    const float* b2  = (const float*)d_in[6];
    const float* w3  = (const float*)d_in[7];
    const float* b3  = (const float*)d_in[8];
    const float* rnd = (const float*)d_in[9];
    const int*   lid = (const int*)d_in[10];
    float* out = (float*)d_out;

    char* ws = (char*)d_ws;
    f16* xln = (f16*)(ws + (0ull << 20));                 // 1 MB  [2048,256] f16
    f16* h1  = (f16*)(ws + (1ull << 20));                 // 1 MB
    f16* h2  = (f16*)(ws + (2ull << 20));                 // 1 MB
    f16* me  = (f16*)(ws + (3ull << 20));                 // 1 MB
    f16* w1t = (f16*)(ws + (4ull << 20));                 // 128 KB each
    f16* w2t = (f16*)(ws + (4ull << 20) + (1ull << 17));
    f16* w3t = (f16*)(ws + (4ull << 20) + (2ull << 17));
    float* attn   = (float*)(ws + (5ull << 20));          // 8 MB  [2,1024,1024] f32
    float* rowsum = (float*)(ws + (13ull << 20));         // 8 KB  [2048] f32

    // weight transposes (f32 -> f16, n-major)
    k_wt<<<256, 256, 0, stream>>>(w1, w1t);
    k_wt<<<256, 256, 0, stream>>>(w2, w2t);
    k_wt<<<256, 256, 0, stream>>>(w3, w3t);

    // layernorm
    k_ln<<<BATCH * QLEN, 256, 0, stream>>>(tgt, lnw, lnb, xln);

    // 3-layer MLP via WMMA (2048 tiles -> 512 blocks of 4 waves)
    k_gemm_mlp<<<512, 128, 0, stream>>>(xln, w1t, b1, h1, 1);
    k_gemm_mlp<<<512, 128, 0, stream>>>(h1,  w2t, b2, h2, 1);
    k_gemm_mlp<<<512, 128, 0, stream>>>(h2,  w3t, b3, me, 0);

    // attn = sigmoid(me @ me^T / 16) via WMMA (8192 tiles -> 2048 blocks)
    k_scores<<<2048, 128, 0, stream>>>(me, attn);

    // Boltzmann numerator row sums (b128 loads)
    k_rowsum<<<BATCH * QLEN, 256, 0, stream>>>((const float4*)attn, lid, rowsum);

    // fused tail, 4 columns x 8 heads per thread, all b128 traffic
    k_final<<<(BATCH * QLEN * QLEN / 4) / 256, 256, 0, stream>>>(
        (const float4*)attn, rowsum, (const float4*)rnd, lid, (float4*)out);
}
